// TTT_Layer_55877524521402
// MI455X (gfx1250) — matrix operand order repaired and checked by reading, not compiled, not run on hardware
//
#include <hip/hip_runtime.h>
#include <hip/hip_bf16.h>

// CDNA5 (gfx1250) wave32 TTT layer:
//   z = x @ thetaQ ; h = gelu(z@W1+b1) ; y = h@W2+b2 ; out = z + tanh(alpha)*LN(y)
// GEMMs on v_wmma_f32_16x16x32_bf16. Tile staging via global_load_async_to_lds_b128
// (ASYNCcnt), double-buffered so copies overlap the WMMA stream. Weights pre-converted
// once to bf16 *transposed* (N-major, K-contiguous) so staging is pure 16B copies.

typedef __attribute__((ext_vector_type(16))) __bf16 v16bf;
typedef __attribute__((ext_vector_type(8)))  __bf16 v8bf;
typedef __attribute__((ext_vector_type(8)))  float  v8f;

#define BM 128
#define BN 128
#define BK 64

__device__ __forceinline__ __bf16 f2bf(float x) { return (__bf16)x; }

// ---- CDNA5 async global->LDS copy (ASYNCcnt tracked) ----
__device__ __forceinline__ void async_copy16(unsigned lds_off, unsigned long long gaddr) {
    asm volatile("global_load_async_to_lds_b128 %0, %1, off"
                 :: "v"(lds_off), "v"(gaddr) : "memory");
}
__device__ __forceinline__ void wait_async0() {
    asm volatile("s_wait_asynccnt 0" ::: "memory");
}

// -------------------- bf16 convert / transpose-convert --------------------
__global__ void __launch_bounds__(256)
cvt_bf16(const float* __restrict__ in, __bf16* __restrict__ out)
{
    const int i = (blockIdx.x * 256 + threadIdx.x) * 4;
    float4 v = *(const float4*)(in + i);
    union { __bf16 b[4]; uint2 u; } pk;
    pk.b[0] = f2bf(v.x); pk.b[1] = f2bf(v.y); pk.b[2] = f2bf(v.z); pk.b[3] = f2bf(v.w);
    *(uint2*)(out + i) = pk.u;
}

// out[c][r] = bf16(in[r][c]);  in: R x C fp32 row-major, out: C x R bf16 row-major
__global__ void __launch_bounds__(256)
transpose_cvt(const float* __restrict__ in, __bf16* __restrict__ out, int R, int C)
{
    __shared__ float t[32][33];
    const int rb = blockIdx.y * 32, cb = blockIdx.x * 32;
    const int tx = threadIdx.x & 31, ty = threadIdx.x >> 5;  // ty: 0..7
#pragma unroll
    for (int j = 0; j < 32; j += 8)
        t[ty + j][tx] = in[(size_t)(rb + ty + j) * C + cb + tx];
    __syncthreads();
#pragma unroll
    for (int j = 0; j < 32; j += 8)
        out[(size_t)(cb + ty + j) * R + rb + tx] = f2bf(t[tx][ty + j]);
}

// -------------------- WMMA GEMM: C[M,N] = A[M,K] * Bt[N,K]^T --------------------
// EPI: 0 = store fp32 Cf AND bf16 Cb ; 1 = bias + exact GELU -> bf16 Cb ; 2 = bias -> fp32 Cf
template<int N, int K, int EPI>
__global__ void __launch_bounds__(256)
wmma_gemm(const __bf16* __restrict__ A, const __bf16* __restrict__ Bt,
          const float* __restrict__ bias,
          float* __restrict__ Cf, __bf16* __restrict__ Cb)
{
    // double-buffered tiles: [buf][row][k], rows are 128B
    __shared__ __align__(16) __bf16 As[2][BM * BK];
    __shared__ __align__(16) __bf16 Bs[2][BN * BK];

    const int tid   = threadIdx.x;
    const int lane  = tid & 31;
    const int wave  = tid >> 5;
    const int waveM = wave >> 2;          // 0..1  -> 64-row strip
    const int waveN = wave & 3;           // 0..3  -> 32-col strip

    const int bm = blockIdx.y * BM;
    const int bn = blockIdx.x * BN;

    v8f acc[4][2];
#pragma unroll
    for (int i = 0; i < 4; ++i)
#pragma unroll
        for (int j = 0; j < 2; ++j)
            acc[i][j] = (v8f){0.f,0.f,0.f,0.f,0.f,0.f,0.f,0.f};

    // 16B chunks: tile = 128 rows x 128B -> 1024 chunks; 256 threads x 4 chunks
    const unsigned as_base[2] = { (unsigned)(uintptr_t)&As[0][0],
                                  (unsigned)(uintptr_t)&As[1][0] };
    const unsigned bs_base[2] = { (unsigned)(uintptr_t)&Bs[0][0],
                                  (unsigned)(uintptr_t)&Bs[1][0] };
    unsigned long long ag[4], bg[4];
    unsigned           aoff[4], boff[4];
#pragma unroll
    for (int q = 0; q < 4; ++q) {
        const int c    = tid + q * 256;        // chunk id 0..1023
        const int row  = c >> 3;               // 0..127
        const int part = c & 7;                // 16B part within 128B row
        ag[q] = (unsigned long long)(uintptr_t)(A  + (size_t)(bm + row) * K + part * 8);
        bg[q] = (unsigned long long)(uintptr_t)(Bt + (size_t)(bn + row) * K + part * 8);
        aoff[q] = (unsigned)c * 16;
        boff[q] = (unsigned)c * 16;
    }

    // prologue: stage tile 0 into buffer 0
#pragma unroll
    for (int q = 0; q < 4; ++q) async_copy16(as_base[0] + aoff[q], ag[q]);
#pragma unroll
    for (int q = 0; q < 4; ++q) async_copy16(bs_base[0] + boff[q], bg[q]);

    const int l16   = lane & 15;
    const int ahalf = (lane >> 4) * 8;    // A frag: K chunks at +0 and +16
    const int bhalf = (lane >> 4) * 8;    // B frag: K chunks at +0 and +16

    for (int k0 = 0; k0 < K; k0 += BK) {
        const int cur = (k0 / BK) & 1;
        wait_async0();          // current buffer's copies complete (per wave)
        __syncthreads();        // ...and visible to all waves

        // overlap: kick off next tile into the other buffer before computing
        if (k0 + BK < K) {
            const unsigned long long koff = (unsigned long long)(k0 + BK) * 2;
            const int nxt = cur ^ 1;
#pragma unroll
            for (int q = 0; q < 4; ++q) async_copy16(as_base[nxt] + aoff[q], ag[q] + koff);
#pragma unroll
            for (int q = 0; q < 4; ++q) async_copy16(bs_base[nxt] + boff[q], bg[q] + koff);
        }

        const __bf16* __restrict__ Ab = &As[cur][0];
        const __bf16* __restrict__ Bb = &Bs[cur][0];
#pragma unroll
        for (int kk = 0; kk < BK; kk += 32) {
            v16bf afrag[4];
#pragma unroll
            for (int mt = 0; mt < 4; ++mt) {
                const int r = waveM * 64 + mt * 16 + l16;
                v8bf lo = *(const v8bf*)&Ab[r * BK + kk + ahalf];
                v8bf hi = *(const v8bf*)&Ab[r * BK + kk + ahalf + 16];
                ((v8bf*)&afrag[mt])[0] = lo;
                ((v8bf*)&afrag[mt])[1] = hi;
            }
            v16bf bfrag[2];
#pragma unroll
            for (int nt = 0; nt < 2; ++nt) {
                const int cidx = waveN * 32 + nt * 16 + l16;
                v8bf lo = *(const v8bf*)&Bb[cidx * BK + kk + bhalf];
                v8bf hi = *(const v8bf*)&Bb[cidx * BK + kk + bhalf + 16];
                ((v8bf*)&bfrag[nt])[0] = lo;
                ((v8bf*)&bfrag[nt])[1] = hi;
            }
#pragma unroll
            for (int mt = 0; mt < 4; ++mt)
#pragma unroll
                for (int nt = 0; nt < 2; ++nt)
                    acc[mt][nt] = __builtin_amdgcn_wmma_f32_16x16x32_bf16(
                        false, afrag[mt], false, bfrag[nt],
                        (short)0, acc[mt][nt], false, false);
        }
        // NOTE: no trailing barrier needed — each thread's LDS reads are dscnt-waited
        // before its WMMAs, and the next iteration's top barrier orders buffer reuse.
    }

    // ---- epilogue: row = base + (lane>>4)*8 + i, col = lane&15 ----
    const int rhalf = (lane >> 4) * 8;
#pragma unroll
    for (int mt = 0; mt < 4; ++mt) {
#pragma unroll
        for (int nt = 0; nt < 2; ++nt) {
            const int rbase = bm + waveM * 64 + mt * 16 + rhalf;
            const int cbase = bn + waveN * 32 + nt * 16 + l16;
            const size_t base = (size_t)rbase * N + cbase;
            const float bval = (EPI != 0) ? bias[cbase] : 0.0f;
#pragma unroll
            for (int i = 0; i < 8; ++i) {
                const size_t off = base + (size_t)i * N;  // immediate offsets (N is constexpr)
                const float v = acc[mt][nt][i] + bval;
                if (EPI == 0) {
                    Cf[off] = acc[mt][nt][i];
                    Cb[off] = f2bf(acc[mt][nt][i]);
                } else if (EPI == 1) {
                    const float g = 0.5f * v * (1.0f + erff(v * 0.7071067811865476f));
                    Cb[off] = f2bf(g);
                } else {
                    Cf[off] = v;
                }
            }
        }
    }
}

// ---- Row-wise LayerNorm over D=1024 + residual: out = z + tanh(alpha)*ln(y) ----
__global__ void __launch_bounds__(256)
ln_combine(const float* __restrict__ z, const float* __restrict__ y,
           const float* __restrict__ gamma, const float* __restrict__ beta,
           const float* __restrict__ alpha, float* __restrict__ out)
{
    __shared__ float s1[256];
    __shared__ float s2[256];
    const int r   = blockIdx.x;
    const int tid = threadIdx.x;
    const float* yr = y + (size_t)r * 1024;

    float vals[4];
    float sum = 0.f, sq = 0.f;
#pragma unroll
    for (int j = 0; j < 4; ++j) {
        float v = yr[tid + j * 256];
        vals[j] = v;
        sum += v;
        sq  += v * v;
    }
    s1[tid] = sum; s2[tid] = sq;
    __syncthreads();
#pragma unroll
    for (int s = 128; s > 0; s >>= 1) {
        if (tid < s) { s1[tid] += s1[tid + s]; s2[tid] += s2[tid + s]; }
        __syncthreads();
    }
    const float mu  = s1[0] * (1.0f / 1024.0f);
    const float var = s2[0] * (1.0f / 1024.0f) - mu * mu;   // biased, as torch LN
    const float inv = rsqrtf(var + 1e-5f);

#pragma unroll
    for (int j = 0; j < 4; ++j) {
        const int d = tid + j * 256;
        const float ln = (vals[j] - mu) * inv * gamma[d] + beta[d];
        const size_t off = (size_t)r * 1024 + d;
        out[off] = z[off] + tanhf(alpha[d]) * ln;
    }
}

extern "C" void kernel_launch(void* const* d_in, const int* in_sizes, int n_in,
                              void* d_out, int out_size, void* d_ws, size_t ws_size,
                              hipStream_t stream)
{
    const float* x      = (const float*)d_in[0];
    const float* thetaQ = (const float*)d_in[1];
    const float* W1     = (const float*)d_in[2];
    const float* b1     = (const float*)d_in[3];
    const float* W2     = (const float*)d_in[4];
    const float* b2     = (const float*)d_in[5];
    const float* gamma  = (const float*)d_in[6];
    const float* beta   = (const float*)d_in[7];
    const float* alpha  = (const float*)d_in[8];

    const int M  = 2 * 8 * 512;   // 8192 token rows
    const int Dd = 1024;
    const int Hh = 4096;

    // workspace layout (bytes)
    char* ws = (char*)d_ws;
    float*  z   = (float*)(ws);                                    // 32 MB
    __bf16* zb  = (__bf16*)(ws + (size_t)32  * 1024 * 1024);       // 16 MB
    __bf16* h   = (__bf16*)(ws + (size_t)48  * 1024 * 1024);       // 64 MB
    float*  y   = (float*) (ws + (size_t)112 * 1024 * 1024);       // 32 MB
    __bf16* xb  = (__bf16*)(ws + (size_t)144 * 1024 * 1024);       // 16 MB
    __bf16* Qt  = (__bf16*)(ws + (size_t)160 * 1024 * 1024);       //  2 MB
    __bf16* W1t = (__bf16*)(ws + (size_t)162 * 1024 * 1024);       //  8 MB
    __bf16* W2t = (__bf16*)(ws + (size_t)170 * 1024 * 1024);       //  8 MB

    dim3 blk(256);

    // one-time conversions: x -> bf16 ; weights -> bf16 transposed (N-major, K-contig)
    cvt_bf16<<<dim3((M * Dd) / 1024), blk, 0, stream>>>(x, xb);
    transpose_cvt<<<dim3(Dd / 32, Dd / 32), blk, 0, stream>>>(thetaQ, Qt, Dd, Dd);
    transpose_cvt<<<dim3(Hh / 32, Dd / 32), blk, 0, stream>>>(W1, W1t, Dd, Hh);
    transpose_cvt<<<dim3(Dd / 32, Hh / 32), blk, 0, stream>>>(W2, W2t, Hh, Dd);

    // GEMM1: z (fp32) & zb (bf16) = x @ thetaQ
    wmma_gemm<1024, 1024, 0><<<dim3(Dd / BN, M / BM), blk, 0, stream>>>(
        xb, Qt, nullptr, z, zb);
    // GEMM2: h = gelu(z @ W1 + b1), bf16
    wmma_gemm<4096, 1024, 1><<<dim3(Hh / BN, M / BM), blk, 0, stream>>>(
        zb, W1t, b1, nullptr, h);
    // GEMM3: y = h @ W2 + b2, fp32
    wmma_gemm<1024, 4096, 2><<<dim3(Dd / BN, M / BM), blk, 0, stream>>>(
        h, W2t, b2, y, nullptr);

    // LayerNorm + residual combine
    ln_combine<<<dim3(M), blk, 0, stream>>>(z, y, gamma, beta, alpha, (float*)d_out);
}
